// CrossClipMerging_12266426598092
// MI455X (gfx1250) — matrix operand order repaired and checked by reading, not compiled, not run on hardware
//
#include <hip/hip_runtime.h>

// Problem shape from the reference (setup_inputs): B=16, N=2048, D=256, fp32.
#define BB 16
#define NN 2048
#define DD 256

typedef __attribute__((ext_vector_type(16))) __bf16        v16bf;
typedef __attribute__((ext_vector_type(8)))  float         v8f;
typedef __attribute__((ext_vector_type(4)))  unsigned int  v4u;
typedef __attribute__((ext_vector_type(4)))  float         v4f;
typedef __attribute__((address_space(3)))    unsigned short lds_ushort_t;

union FragBF {
    v16bf v;
    v4u   u[2];
};

// Round-to-nearest-even f32 -> bf16 (inputs are finite, |x|<=1 after normalize).
__device__ __forceinline__ unsigned short f32_to_bf16(float f) {
    unsigned int u = __float_as_uint(f);
    u += 0x7FFFu + ((u >> 16) & 1u);
    return (unsigned short)(u >> 16);
}

// Monotone (value, col) -> u64 key. max(key) == (max value, lowest col on ties),
// matching jnp.argmax tie-breaking. Negative floats handled via sign-flip trick.
__device__ __forceinline__ unsigned long long packKey(float v, unsigned int col) {
    unsigned int u = __float_as_uint(v);
    u = (u & 0x80000000u) ? ~u : (u | 0x80000000u);
    return ((unsigned long long)u << 32) | (unsigned long long)(~col);
}

// CDNA5 async DMA: global -> LDS, 16 bytes, tracked by ASYNCcnt (no VGPR data).
__device__ __forceinline__ void async_copy_b128(unsigned lds_byte_off,
                                                const unsigned short* gaddr) {
    asm volatile("global_load_async_to_lds_b128 %0, %1, off"
                 :: "v"(lds_byte_off), "v"(gaddr)
                 : "memory");
}

__device__ __forceinline__ void wait_asynccnt0() {
    asm volatile("s_wait_asynccnt 0x0" ::: "memory");
}

// ---------------------------------------------------------------------------
// Kernel 1: L2-normalize rows (f32 -> bf16 scratch) and zero the argmax keys.
// One wave per row: 32 lanes * 8 floats = 256 = D. gridDim.y selects clip1/2.
// ---------------------------------------------------------------------------
__global__ void normalize_kernel(const float* __restrict__ c1,
                                 const float* __restrict__ c2,
                                 unsigned short* __restrict__ n1,
                                 unsigned short* __restrict__ n2,
                                 unsigned long long* __restrict__ keys) {
    const float*    src = blockIdx.y ? c2 : c1;
    unsigned short* dst = blockIdx.y ? n2 : n1;
    const int wave = threadIdx.x >> 5;
    const int lane = threadIdx.x & 31;
    const int row  = blockIdx.x * 8 + wave;           // 4096 blocks * 8 rows

    const float* p = src + (size_t)row * DD + lane * 8;
    v4f a = *(const v4f*)p;
    v4f b = *(const v4f*)(p + 4);
    float s = a[0]*a[0] + a[1]*a[1] + a[2]*a[2] + a[3]*a[3]
            + b[0]*b[0] + b[1]*b[1] + b[2]*b[2] + b[3]*b[3];
#pragma unroll
    for (int off = 16; off; off >>= 1) s += __shfl_xor(s, off, 32);

    const float scale = 1.0f / fmaxf(sqrtf(s), 1e-8f);  // torch eps clamp
    float e[8] = {a[0], a[1], a[2], a[3], b[0], b[1], b[2], b[3]};
    v4u w;
#pragma unroll
    for (int i = 0; i < 4; ++i) {
        unsigned int lo = f32_to_bf16(e[2*i]     * scale);
        unsigned int hi = f32_to_bf16(e[2*i + 1] * scale);
        w[i] = lo | (hi << 16);
    }
    *(v4u*)(dst + (size_t)row * DD + lane * 8) = w;

    if (blockIdx.y == 0 && lane == 0) keys[row] = 0ull;  // init (ws is poisoned)
}

// ---------------------------------------------------------------------------
// Kernel 2: bf16 WMMA GEMM (sim = n1 . n2^T per batch) fused with row argmax.
// Block = 8 waves = one 128-row M superblock x one 128-col N strip.
// The block async-DMAs the shared 64KB B strip into LDS once (8x L2 traffic
// reduction); each wave then computes a 16x128 tile: 8 accumulators, 64 WMMAs,
// B fragments via ds_load_b128 broadcast, A fragments from global.
// ---------------------------------------------------------------------------
__global__ void gemm_argmax_kernel(const unsigned short* __restrict__ n1,
                                   const unsigned short* __restrict__ n2,
                                   unsigned long long* __restrict__ keys) {
    __shared__ unsigned short sB[128 * DD];   // 64 KB: B strip, row-major per col

    const int tid   = threadIdx.x;
    const int wave  = tid >> 5;
    const int lane  = tid & 31;
    const int half  = lane >> 4;              // 0 or 1 (lane group)
    const int l     = lane & 15;

    const int blk    = blockIdx.x;            // 4096 blocks
    const int b      = blk >> 8;              // batch
    const int msuper = (blk >> 4) & 15;       // 128-row superblock in batch
    const int strip  = blk & 15;              // 128-col N strip
    const int mloc   = msuper * 128 + wave * 16;
    const int nbase  = strip * 128;

    const unsigned short* Abase = n1 + (size_t)b * NN * DD;
    const unsigned short* Bbase = n2 + (size_t)b * NN * DD;

    // --- Cooperative async copy of the B strip: 4096 16B chunks, coalesced ---
    {
#pragma unroll
        for (int c = 0; c < 16; ++c) {
            const int chunk = tid + c * 256;          // 0..4095
            const int col   = chunk >> 5;             // 512B (32 chunks) per col
            const int elt   = (chunk & 31) * 8;       // bf16 element offset
            const unsigned short* g = Bbase + (size_t)(nbase + col) * DD + elt;
            const unsigned lds_off =
                (unsigned)(unsigned long long)(lds_ushort_t*)(sB + col * DD + elt);
            async_copy_b128(lds_off, g);
        }
        wait_asynccnt0();     // this wave's DMAs landed in LDS
        __syncthreads();      // all waves' DMAs landed
    }

    const unsigned short* arow = Abase + (size_t)(mloc + l) * DD;

    v8f acc[8];
#pragma unroll
    for (int t = 0; t < 8; ++t) acc[t] = (v8f){};

    // K loop: D=256 in steps of 32.
#pragma unroll
    for (int kb = 0; kb < DD; kb += 32) {
        // A layout (16-bit 16x32): lanes 0-15 hold K 0..7 (v0-3) and 16..23
        // (v4-7); lanes 16-31 hold K 8..15 and 24..31. Row M = lane%16.
        FragBF a;
        a.u[0] = *(const v4u*)(arow + kb + half * 8);
        a.u[1] = *(const v4u*)(arow + kb + 16 + half * 8);

#pragma unroll
        for (int t = 0; t < 8; ++t) {
            // B layout (same K order as A): lanes 0-15 hold K kb..kb+7 and
            // kb+16..kb+23, lanes 16-31 hold K kb+8..kb+15 and kb+24..kb+31,
            // column N = lane%16. Columns live contiguous
            // in LDS -> two ds_load_b128, broadcast across all 8 waves.
            const unsigned short* bp =
                sB + (nbase ? 0 : 0) + (t * 16 + l) * DD + kb + half * 8;
            FragBF bb;
            bb.u[0] = *(const v4u*)(bp);
            bb.u[1] = *(const v4u*)(bp + 16);

            acc[t] = __builtin_amdgcn_wmma_f32_16x16x32_bf16(
                false, a.v, false, bb.v, (short)0, acc[t], false, false);
        }
    }

    // C layout: VGPR c, lanes 0-15 -> M=c, N=lane; lanes 16-31 -> M=c+8,
    // N=lane-16. Per lane: component c holds row (c + 8*half), col nbase+t*16+l.
    const int rowbase = b * NN + mloc;
#pragma unroll
    for (int c = 0; c < 8; ++c) {
        unsigned long long best = 0ull;
#pragma unroll
        for (int t = 0; t < 8; ++t) {
            unsigned long long k = packKey(acc[t][c],
                                           (unsigned int)(nbase + t * 16 + l));
            best = (k > best) ? k : best;
        }
        // Reduce across the 16 lanes of this half (xor<16 stays in-half).
#pragma unroll
        for (int off = 8; off; off >>= 1) {
            unsigned long long o = __shfl_xor(best, off, 32);
            best = (o > best) ? o : best;
        }
        if (l == 0) {
            atomicMax(&keys[rowbase + c + half * 8], best);  // global_atomic_max_u64
        }
    }
}

// ---------------------------------------------------------------------------
// Kernel 3: gather winners and average in exact f32.
// ---------------------------------------------------------------------------
__global__ void gather_kernel(const float* __restrict__ c1,
                              const float* __restrict__ c2,
                              const unsigned long long* __restrict__ keys,
                              float* __restrict__ out) {
    const int gid = blockIdx.x * blockDim.x + threadIdx.x;  // 8.39M threads
    const int row = gid >> 8;        // D = 256
    const int d   = gid & 255;
    const int b   = row >> 11;       // N = 2048
    const unsigned int i = ~(unsigned int)(keys[row] & 0xFFFFFFFFull);
    const size_t s = ((size_t)b * NN + i) * DD + d;
    out[gid] = 0.5f * (c1[s] + c2[s]);
}

extern "C" void kernel_launch(void* const* d_in, const int* in_sizes, int n_in,
                              void* d_out, int out_size, void* d_ws, size_t ws_size,
                              hipStream_t stream) {
    const float* c1 = (const float*)d_in[0];
    const float* c2 = (const float*)d_in[1];
    float* out = (float*)d_out;

    // Workspace layout: bf16 n1 (16.8MB) | bf16 n2 (16.8MB) | u64 keys (256KB)
    unsigned short* n1 = (unsigned short*)d_ws;
    unsigned short* n2 = n1 + (size_t)BB * NN * DD;
    unsigned long long* keys = (unsigned long long*)(n2 + (size_t)BB * NN * DD);

    // 1) Normalize both tensors to bf16; zero the argmax keys.
    dim3 gN(BB * NN / 8, 2);
    normalize_kernel<<<gN, 256, 0, stream>>>(c1, c2, n1, n2, keys);

    // 2) WMMA GEMM + fused argmax:
    //    16 batches * 16 M-superblocks * 16 N-strips = 4096 blocks of 8 waves.
    gemm_argmax_kernel<<<BB * 16 * 16, 256, 0, stream>>>(n1, n2, keys);

    // 3) Gather + average (one thread per output element).
    gather_kernel<<<(BB * NN * DD) / 256, 256, 0, stream>>>(c1, c2, keys, out);
}
